// Attention_15118284882236
// MI455X (gfx1250) — compile-verified, matrix operand order checked by reading, not run on hardware
//
#include <hip/hip_runtime.h>

// Fused scaled-dot-product attention for (B,LQ,LKV,D,DV)=(8,2048,2048,128,128),
// f32 in / f32 out, returning BOTH softmax weights and context.
// Path: f16 operands + v_wmma_f32_16x16x32_f16 (f32 accumulate), 3 streaming
// passes (rowmax / rowsum+stash exp(s-m) as f16 / normalize+write+PV GEMM).
// No LDS memory, no inter-wave sync; each wave owns a 16-query strip.
// Weights/context stores are non-temporal (write-once streams) so L2 keeps
// the f16 P stash + K/V resident for the pass-3 re-read.
// Workspace use: Qh(4MB)+Kh(4MB)+Vt(4MB)+Ph(67MB) ~= 80MB of d_ws.

typedef __attribute__((ext_vector_type(16))) _Float16 v16h;
typedef __attribute__((ext_vector_type(8)))  _Float16 v8h;
typedef __attribute__((ext_vector_type(8)))  float    v8f;
typedef __attribute__((ext_vector_type(4)))  float    v4f;

#define NB   8
#define SLQ  2048
#define SLKV 2048
#define SD   128
#define SDV  128

// Load a 16x32 f16 A/B fragment from a row-major matrix (row stride `ld`
// elements). Lane L: row = L%16; 8 contiguous cols at 8*(L/16), 8 more at
// 16+8*(L/16). Two global_load_b128 per lane.
__device__ __forceinline__ v16h load_ab_16x32(const _Float16* __restrict__ p,
                                              int ld, int lane) {
    const int r = lane & 15;
    const int h = lane >> 4;
    const _Float16* base = p + (size_t)r * ld + h * 8;
    v8h lo = *reinterpret_cast<const v8h*>(base);
    v8h hi = *reinterpret_cast<const v8h*>(base + 16);
    v16h out;
#pragma unroll
    for (int i = 0; i < 8; ++i) { out[i] = lo[i]; out[i + 8] = hi[i]; }
    return out;
}

__global__ void cvt_f32_to_f16(const float* __restrict__ in,
                               _Float16* __restrict__ out, int n) {
    int i = blockIdx.x * blockDim.x + threadIdx.x;
    if (i < n) out[i] = (_Float16)in[i];
}

// Vt[b][dv][kv] = (f16) V[b][kv][dv]  (writes coalesced; reads L2-resident)
__global__ void transpose_v_f16(const float* __restrict__ V,
                                _Float16* __restrict__ Vt) {
    int i = blockIdx.x * blockDim.x + threadIdx.x;   // NB*SDV*SLKV elements
    int kv = i & (SLKV - 1);
    int dv = (i >> 11) & (SDV - 1);
    int b  = i >> 18;
    Vt[i] = (_Float16)V[((size_t)(b * SLKV + kv)) * SDV + dv];
}

__global__ __launch_bounds__(256, 1) void attn_fused(
    const _Float16* __restrict__ Qh,
    const _Float16* __restrict__ Kh,
    const _Float16* __restrict__ Vt,
    _Float16* __restrict__ Ph,
    float* __restrict__ Wout,
    float* __restrict__ Cout)
{
    const int lane = threadIdx.x & 31;
    const int wave = threadIdx.x >> 5;
    const int gw   = blockIdx.x * 8 + wave;   // 1024 waves total
    const int b    = gw >> 7;                 // 128 q-tiles per batch
    const int q0   = (gw & 127) << 4;         // 16 queries per wave

    const int r = lane & 15;
    const int h = lane >> 4;

    const float slog2e = 0.08838834764831845f * 1.4426950408889634f; // scale*log2(e)

    // ---- Q fragments (A operand), register-resident for both QK^T passes
    v16h aq[4];
#pragma unroll
    for (int c = 0; c < 4; ++c)
        aq[c] = load_ab_16x32(Qh + (size_t)(b * SLQ + q0) * SD + c * 32, SD, lane);

    const _Float16* Kb = Kh + (size_t)b * SLKV * SD;

    // ================= pass 1: per-lane partial row max (reduce deferred)
    float m[8];
#pragma unroll
    for (int v = 0; v < 8; ++v) m[v] = -3.0e38f;

    for (int n = 0; n < SLKV / 16; ++n) {
        v8f s = {};
#pragma unroll
        for (int c = 0; c < 4; ++c) {
            v16h bk = load_ab_16x32(Kb + (size_t)(n * 16) * SD + c * 32, SD, lane);
            s = __builtin_amdgcn_wmma_f32_16x16x32_f16(false, aq[c], false, bk,
                                                       (short)0, s, false, false);
        }
#pragma unroll
        for (int v = 0; v < 8; ++v) m[v] = fmaxf(m[v], s[v]);
    }
    // cross-lane max within each 16-lane half (xor masks stay inside half)
#pragma unroll
    for (int v = 0; v < 8; ++v) {
#pragma unroll
        for (int off = 1; off < 16; off <<= 1)
            m[v] = fmaxf(m[v], __shfl_xor(m[v], off, 32));
    }

    // ================= pass 2: p = exp2((s-m)*scale*log2e); row sums; stash p
    float l[8];
#pragma unroll
    for (int v = 0; v < 8; ++v) l[v] = 0.0f;

    _Float16* Pb = Ph + (size_t)(b * SLQ + q0) * SLKV;
    for (int n = 0; n < SLKV / 16; ++n) {
        v8f s = {};
#pragma unroll
        for (int c = 0; c < 4; ++c) {
            v16h bk = load_ab_16x32(Kb + (size_t)(n * 16) * SD + c * 32, SD, lane);
            s = __builtin_amdgcn_wmma_f32_16x16x32_f16(false, aq[c], false, bk,
                                                       (short)0, s, false, false);
        }
#pragma unroll
        for (int v = 0; v < 8; ++v) {
            float p = exp2f((s[v] - m[v]) * slog2e);
            l[v] += p;
            // C/D layout element (M=v+8h, N=r) -> row-major stash (default TH:
            // we WANT these dirty-resident in L2 for the pass-3 re-read)
            Pb[(size_t)(v + 8 * h) * SLKV + n * 16 + r] = (_Float16)p;
        }
    }
#pragma unroll
    for (int v = 0; v < 8; ++v) {
#pragma unroll
        for (int off = 1; off < 16; off <<= 1)
            l[v] += __shfl_xor(l[v], off, 32);
    }
    float rl[8];
#pragma unroll
    for (int v = 0; v < 8; ++v) rl[v] = 1.0f / l[v];

    // Redistribute: pass-3 A-layout lane owns row (lane&15); rl[v] in this
    // lane covers rows v+8h. Select own slot (lane&7), then fetch from the
    // half that actually owns the wanted row.
    float rlOwn = rl[0];
#pragma unroll
    for (int v = 1; v < 8; ++v) rlOwn = ((lane & 7) == v) ? rl[v] : rlOwn;
    const int srcLane = ((lane >> 3) & 1) * 16 + (lane & 7);
    const float rlA = __shfl(rlOwn, srcLane, 32);

    // ================= pass 3: normalize, write weights, context = P.V
    v8f cacc[8];
#pragma unroll
    for (int t = 0; t < 8; ++t) { v8f z = {}; cacc[t] = z; }

    float* Wb = Wout + (size_t)(b * SLQ + q0 + r) * SLKV; // this lane's q row
    const _Float16* Vb = Vt + (size_t)b * SDV * SLKV;

    for (int g = 0; g < SLKV / 32; ++g) {
        v16h pr = load_ab_16x32(Pb + g * 32, SLKV, lane);  // A-layout, contiguous

        v16h an;
        v4f w0, w1, w2, w3;
#pragma unroll
        for (int i = 0; i < 4; ++i) {
            float a0 = (float)pr[i]      * rlA;   // cols g*32 + 8h + i
            float a1 = (float)pr[i + 4]  * rlA;   // cols g*32 + 8h + 4 + i
            float b0 = (float)pr[i + 8]  * rlA;   // cols g*32 + 16 + 8h + i
            float b1 = (float)pr[i + 12] * rlA;
            w0[i] = a0; w1[i] = a1; w2[i] = b0; w3[i] = b1;
            an[i] = (_Float16)a0; an[i + 4]  = (_Float16)a1;
            an[i + 8] = (_Float16)b0; an[i + 12] = (_Float16)b1;
        }
        // Weights are a write-once 134MB stream: non-temporal so they do not
        // evict the L2-resident P stash / K / V.
        float* wp = Wb + g * 32 + 8 * h;
        __builtin_nontemporal_store(w0, reinterpret_cast<v4f*>(wp));
        __builtin_nontemporal_store(w1, reinterpret_cast<v4f*>(wp + 4));
        __builtin_nontemporal_store(w2, reinterpret_cast<v4f*>(wp + 16));
        __builtin_nontemporal_store(w3, reinterpret_cast<v4f*>(wp + 20));

#pragma unroll
        for (int t = 0; t < 8; ++t) {
            // B[k][n] = V[kv0+k][t*16+n] = Vt[t*16+n][kv0+k] -> contiguous loads
            v16h bv = load_ab_16x32(Vb + (size_t)(t * 16) * SLKV + g * 32,
                                    SLKV, lane);
            cacc[t] = __builtin_amdgcn_wmma_f32_16x16x32_f16(false, an, false, bv,
                                                             (short)0, cacc[t],
                                                             false, false);
        }
    }

    // ---- write context (C/D layout element (M=v+8h, N=t*16+r)), write-once
    float* Cb = Cout + (size_t)(b * SLQ + q0) * SDV;
#pragma unroll
    for (int t = 0; t < 8; ++t)
#pragma unroll
        for (int v = 0; v < 8; ++v)
            __builtin_nontemporal_store(
                cacc[t][v], Cb + (size_t)(v + 8 * h) * SDV + t * 16 + r);
}

extern "C" void kernel_launch(void* const* d_in, const int* in_sizes, int n_in,
                              void* d_out, int out_size, void* d_ws, size_t ws_size,
                              hipStream_t stream) {
    (void)in_sizes; (void)n_in; (void)out_size; (void)ws_size;
    const float* Q = (const float*)d_in[0];
    const float* K = (const float*)d_in[1];
    const float* V = (const float*)d_in[2];

    float* Wout = (float*)d_out;                          // (B,LQ,LKV) f32
    float* Cout = Wout + (size_t)NB * SLQ * SLKV;         // (B,LQ,DV)  f32

    _Float16* ws = (_Float16*)d_ws;
    const size_t nQ = (size_t)NB * SLQ  * SD;             // 2M halves
    const size_t nK = (size_t)NB * SLKV * SD;             // 2M halves
    const size_t nV = (size_t)NB * SDV  * SLKV;           // 2M halves
    _Float16* Qh = ws;
    _Float16* Kh = Qh + nQ;
    _Float16* Vt = Kh + nK;
    _Float16* Ph = Vt + nV;                               // 33.5M halves

    cvt_f32_to_f16<<<(int)((nQ + 255) / 256), 256, 0, stream>>>(Q, Qh, (int)nQ);
    cvt_f32_to_f16<<<(int)((nK + 255) / 256), 256, 0, stream>>>(K, Kh, (int)nK);
    transpose_v_f16<<<(int)((nV + 255) / 256), 256, 0, stream>>>(V, Vt);

    // 1024 waves: 8 batches x 128 q-tiles of 16 rows; 8 waves per block
    attn_fused<<<128, 256, 0, stream>>>(Qh, Kh, Vt, Ph, Wout, Cout);
}
